// Transformer_58909771432482
// MI455X (gfx1250) — matrix operand order repaired and checked by reading, not compiled, not run on hardware
//
#include <hip/hip_runtime.h>

// ---------------- problem constants ----------------
#define DIMV 768
#define DEPTH 6
#define HEADS 12
#define DHEAD 64
#define MLPD 3072
#define BATCH 8
#define SEQ 512
#define TOKENS (BATCH * SEQ)   /* 4096 */
#define INNER 768
#define SCALE_QK 0.125f
#define EPS_LN 1e-5f

typedef _Float16 half_t;
typedef _Float16 v8h  __attribute__((ext_vector_type(8)));
typedef _Float16 v16h __attribute__((ext_vector_type(16)));
typedef float    v8f  __attribute__((ext_vector_type(8)));
typedef unsigned int v4u __attribute__((ext_vector_type(4)));
typedef unsigned int v8u __attribute__((ext_vector_type(8)));

// =====================================================================
// TDM: async 2D tile load global -> LDS with row padding.
// D# per CDNA5 ISA:
//  group0: [1:0]=count=1, [63:32]=lds byte addr, [120:64]=global addr,
//          [127:126]=type=2
//  group1: data_size=2B, pad_enable, pad_interval=32 DWORDs (code 4),
//          pad_amount=4 DWORDs (code 3) -> LDS row pitch = KT+8 halves,
//          tensor/tile dims + dim0 stride
// Issued per-wave; completion via TENSORcnt.
// =====================================================================
__device__ __forceinline__ void tdm_load_2d(const void* gaddr, unsigned int lds_byte,
                                            unsigned int tile_k, unsigned int tile_rows,
                                            unsigned long long stride_elems)
{
  unsigned long long ga = (unsigned long long)(uintptr_t)gaddr;
  v4u g0;
  g0[0] = 1u;                                       // count=1, user desc
  g0[1] = lds_byte;                                 // lds_addr (bytes)
  g0[2] = (unsigned int)(ga & 0xFFFFFFFFu);         // global_addr[31:0]
  g0[3] = (unsigned int)((ga >> 32) & 0x01FFFFFFu)  // global_addr[56:32]
        | (2u << 30);                               // type=2 (image)
  v8u g1;
  g1[0] = (1u << 16)      // data_size = 2 bytes
        | (1u << 20)      // pad_enable
        | (4u << 22)      // pad_interval: 32 DWORDs (=128B = one 64-half row)
        | (3u << 25);     // pad_amount: 4 DWORDs (=8 halves)
  g1[1] = (tile_k & 0xFFFFu) << 16;                          // tensor_dim0 lo
  g1[2] = (tile_k >> 16) | ((tile_rows & 0xFFFFu) << 16);    // td0 hi | tensor_dim1 lo
  g1[3] = (tile_rows >> 16) | (tile_k << 16);                // td1 hi | tile_dim0
  g1[4] = tile_rows & 0xFFFFu;                               // tile_dim1 (tile_dim2=0)
  g1[5] = (unsigned int)(stride_elems & 0xFFFFFFFFu);        // dim0 stride lo
  g1[6] = (unsigned int)((stride_elems >> 32) & 0xFFFFu);    // dim0 stride hi
  g1[7] = 0u;
  v4u gz; gz[0] = 0u; gz[1] = 0u; gz[2] = 0u; gz[3] = 0u;    // groups 2/3: unused dims
  asm volatile("tensor_load_to_lds %0, %1, %2, %3"
               :: "s"(g0), "s"(g1), "s"(gz), "s"(gz)
               : "memory");
}

// =====================================================================
// WMMA GEMM:  C[M,N] = epilogue( A[M,K] * B^T ),  A:[M][K] lda, B:[N][K] ldb
// f16 in, f32 accumulate. Double-buffered LDS tiles filled by TDM
// (wave0 -> A tile, wave1 -> B tile), K-step 64 (2 wmma slices).
//   EPI=0: Ch = (half)(acc * alpha)
//   EPI=1: Cf = acc + bias[col]
//   EPI=2: Ch = (half) gelu_exact(acc + bias[col])
// Batched over blockIdx.z = zb*Hn+zh with separate (batch, head) strides.
// Block: 256 threads = 8 waves arranged 4(M) x 2(N); tile BM=128 x BN.
// =====================================================================
template <int BN, int EPI>
__global__ __launch_bounds__(256) void gemm_wmma_tdm(
    const half_t* __restrict__ A, const half_t* __restrict__ B,
    half_t* __restrict__ Ch, float* __restrict__ Cf,
    const float* __restrict__ bias,
    int K, int lda, int ldb, int ldc,
    long aSB, long aSH, long bSB, long bSH, long cSB, long cSH,
    int Hn, float alpha)
{
  constexpr int BM  = 128;
  constexpr int KT  = 64;
  constexpr int LDP = KT + 8;          // padded pitch (matches TDM pad descriptor)
  constexpr int ASZ = BM * LDP;        // halves
  constexpr int BSZ = BN * LDP;
  constexpr int WN  = BN / 2;          // waves: 4 (M) x 2 (N)
  constexpr int NT  = WN / 16;

  __shared__ half_t smem[2][ASZ + BSZ];

  const int z  = blockIdx.z;
  const int zb = z / Hn;
  const int zh = z - zb * Hn;
  A += (long)zb * aSB + (long)zh * aSH;
  B += (long)zb * bSB + (long)zh * bSH;
  const long coff = (long)zb * cSB + (long)zh * cSH;

  const int tid    = threadIdx.x;
  const int wave   = tid >> 5;
  const int lane   = tid & 31;
  const int wave_m = wave & 3;
  const int wave_n = wave >> 2;
  const int l16    = lane & 15;
  const int hi     = lane >> 4;
  const int kA     = hi ? 8  : 0;      // A frag: lanes 16-31 hold K=8..15,24..31
  const int kB     = hi ? 8  : 0;      // B frag: lanes 16-31 hold K=8..15,24..31

  const long bm = (long)blockIdx.y * BM;
  const long bn = (long)blockIdx.x * BN;
  const half_t* Abase = A + bm * (long)lda;
  const half_t* Bbase = B + bn * (long)ldb;
  const int NK = K / KT;

  // per-stage LDS byte offsets for the TDM descriptors
  const unsigned int ldsA0 = (unsigned int)(uintptr_t)&smem[0][0];
  const unsigned int ldsA1 = (unsigned int)(uintptr_t)&smem[1][0];
  const unsigned int ldsB0 = (unsigned int)(uintptr_t)&smem[0][ASZ];
  const unsigned int ldsB1 = (unsigned int)(uintptr_t)&smem[1][ASZ];

  // prologue: request tile 0
  if (wave == 0) tdm_load_2d(Abase, ldsA0, KT, BM, (unsigned long long)lda);
  if (wave == 1) tdm_load_2d(Bbase, ldsB0, KT, BN, (unsigned long long)ldb);

  v8f acc[2][NT] = {};

  for (int kt = 0; kt < NK; ++kt) {
    const int st = kt & 1;
    if (wave < 2) __builtin_amdgcn_s_wait_tensorcnt(0);   // tile kt resident
    __syncthreads();
    // overlap: request tile kt+1 into the other stage while computing kt
    if (kt + 1 < NK) {
      if (wave == 0)
        tdm_load_2d(Abase + (long)(kt + 1) * KT, st ? ldsA0 : ldsA1,
                    KT, BM, (unsigned long long)lda);
      if (wave == 1)
        tdm_load_2d(Bbase + (long)(kt + 1) * KT, st ? ldsB0 : ldsB1,
                    KT, BN, (unsigned long long)ldb);
    }

    const half_t (*As)[LDP] = (const half_t(*)[LDP])&smem[st][0];
    const half_t (*Bs)[LDP] = (const half_t(*)[LDP])&smem[st][ASZ];

#pragma unroll
    for (int ks = 0; ks < 2; ++ks) {           // two K=32 slices per stage
      v16h af[2];
#pragma unroll
      for (int mt = 0; mt < 2; ++mt) {
        int row = wave_m * 32 + mt * 16 + l16;
        v8h lo = *(const v8h*)&As[row][ks * 32 + kA];
        v8h h8 = *(const v8h*)&As[row][ks * 32 + kA + 16];
        af[mt] = __builtin_shufflevector(lo, h8, 0,1,2,3,4,5,6,7,8,9,10,11,12,13,14,15);
      }
      v16h bf[NT];
#pragma unroll
      for (int nt = 0; nt < NT; ++nt) {
        int col = wave_n * WN + nt * 16 + l16;
        v8h lo = *(const v8h*)&Bs[col][ks * 32 + kB];
        v8h h8 = *(const v8h*)&Bs[col][ks * 32 + kB + 16];
        bf[nt] = __builtin_shufflevector(lo, h8, 0,1,2,3,4,5,6,7,8,9,10,11,12,13,14,15);
      }
#pragma unroll
      for (int mt = 0; mt < 2; ++mt)
#pragma unroll
        for (int nt = 0; nt < NT; ++nt)
          acc[mt][nt] = __builtin_amdgcn_wmma_f32_16x16x32_f16(
              false, af[mt], false, bf[nt], (short)0, acc[mt][nt], false, false);
    }
    __syncthreads();   // all waves done with stage st before it is refilled
  }

  // ---- epilogue (C layout: VGPR v -> M = v + 8*hi, N = l16) ----
#pragma unroll
  for (int mt = 0; mt < 2; ++mt) {
#pragma unroll
    for (int nt = 0; nt < NT; ++nt) {
      long col   = bn + wave_n * WN + nt * 16 + l16;
      long rbase = bm + wave_m * 32 + mt * 16 + hi * 8;
#pragma unroll
      for (int v = 0; v < 8; ++v) {
        long off  = coff + (rbase + v) * (long)ldc + col;
        float val = acc[mt][nt][v];
        if (EPI == 0) {
          Ch[off] = (half_t)(val * alpha);
        } else if (EPI == 1) {
          Cf[off] = val + bias[col];
        } else {
          float t = val + bias[col];
          Ch[off] = (half_t)(0.5f * t * (1.0f + erff(t * 0.70710678118654752f)));
        }
      }
    }
  }
}

// =====================================================================
// softmax over row of 512 f16 scores, then multiply by f32 mask, in place
// =====================================================================
__global__ __launch_bounds__(256) void softmax_mask_kernel(
    half_t* __restrict__ S, const float* __restrict__ mask)
{
  const int r   = blockIdx.x;                 // (b,h,i) flat
  const int b   = r / (HEADS * SEQ);
  const int i   = r % SEQ;
  half_t* row   = S + (long)r * SEQ;
  const float* mrow = mask + (long)b * SEQ * SEQ + (long)i * SEQ;
  const int tid = threadIdx.x;

  __shared__ float red[256];
  float v0 = (float)row[tid];
  float v1 = (float)row[tid + 256];

  red[tid] = fmaxf(v0, v1);
  __syncthreads();
  for (int s = 128; s > 0; s >>= 1) {
    if (tid < s) red[tid] = fmaxf(red[tid], red[tid + s]);
    __syncthreads();
  }
  float mx = red[0];
  __syncthreads();

  float e0 = __expf(v0 - mx);
  float e1 = __expf(v1 - mx);
  red[tid] = e0 + e1;
  __syncthreads();
  for (int s = 128; s > 0; s >>= 1) {
    if (tid < s) red[tid] += red[tid + s];
    __syncthreads();
  }
  float inv = 1.0f / red[0];

  row[tid]       = (half_t)(e0 * inv * mrow[tid]);
  row[tid + 256] = (half_t)(e1 * inv * mrow[tid + 256]);
}

// =====================================================================
// X = LayerNorm(Y)*g + b + X  (post-norm residual), also emit f16 copy
// =====================================================================
__global__ __launch_bounds__(256) void ln_residual_kernel(
    const float* __restrict__ Y, const float* __restrict__ g,
    const float* __restrict__ bta, float* __restrict__ X,
    half_t* __restrict__ Xh)
{
  const int t   = blockIdx.x;
  const int tid = threadIdx.x;
  const float* y = Y + (long)t * DIMV;

  float a0 = y[tid], a1 = y[tid + 256], a2 = y[tid + 512];

  __shared__ float red[256];
  red[tid] = a0 + a1 + a2;
  __syncthreads();
  for (int s = 128; s > 0; s >>= 1) {
    if (tid < s) red[tid] += red[tid + s];
    __syncthreads();
  }
  float mu = red[0] * (1.0f / (float)DIMV);
  __syncthreads();

  float d0 = a0 - mu, d1 = a1 - mu, d2 = a2 - mu;
  red[tid] = d0 * d0 + d1 * d1 + d2 * d2;
  __syncthreads();
  for (int s = 128; s > 0; s >>= 1) {
    if (tid < s) red[tid] += red[tid + s];
    __syncthreads();
  }
  float rstd = rsqrtf(red[0] * (1.0f / (float)DIMV) + EPS_LN);

  long base = (long)t * DIMV;
  float dd[3] = {d0, d1, d2};
#pragma unroll
  for (int j = 0; j < 3; ++j) {
    int c    = tid + j * 256;
    float ln = dd[j] * rstd * g[c] + bta[c];
    float nx = ln + X[base + c];
    X[base + c]  = nx;
    Xh[base + c] = (half_t)nx;
  }
}

// =====================================================================
// f32 -> f16 cast (n multiple of 4)
// =====================================================================
__global__ __launch_bounds__(256) void cast_f32_to_f16(
    const float* __restrict__ s, half_t* __restrict__ d, int n)
{
  int i = (blockIdx.x * 256 + threadIdx.x) * 4;
  if (i < n) {
    float4 v = *(const float4*)(s + i);
    d[i + 0] = (half_t)v.x;
    d[i + 1] = (half_t)v.y;
    d[i + 2] = (half_t)v.z;
    d[i + 3] = (half_t)v.w;
  }
}

// =====================================================================
// tiled transpose-cast: dst[n][k] = (half) src[k][n]   (K, N mult of 32)
// =====================================================================
__global__ __launch_bounds__(256) void transpose_cast_kernel(
    const float* __restrict__ src, half_t* __restrict__ dst, int K, int N)
{
  __shared__ float tile[32][33];
  const int tk = blockIdx.x * 32;
  const int tn = blockIdx.y * 32;
  const int r  = threadIdx.x >> 5;
  const int c  = threadIdx.x & 31;
#pragma unroll
  for (int i = 0; i < 4; ++i)
    tile[r + i * 8][c] = src[(long)(tk + r + i * 8) * N + tn + c];
  __syncthreads();
#pragma unroll
  for (int i = 0; i < 4; ++i)
    dst[(long)(tn + r + i * 8) * K + tk + c] = (half_t)tile[c][r + i * 8];
}

// =====================================================================
// transpose V activations: vT[(b*H+h)*64 + d][j] = kv[b*512+j][768 + h*64 + d]
// grid: (SEQ/32, DHEAD/32, B*H)
// =====================================================================
__global__ __launch_bounds__(256) void transpose_v_kernel(
    const half_t* __restrict__ kv, half_t* __restrict__ vT)
{
  __shared__ half_t tile[32][40];
  const int z  = blockIdx.z;
  const int b  = z / HEADS;
  const int h  = z % HEADS;
  const int tj = blockIdx.x * 32;
  const int td = blockIdx.y * 32;
  const int r  = threadIdx.x >> 5;
  const int c  = threadIdx.x & 31;
  const half_t* src = kv + (long)b * SEQ * 2 * INNER + INNER + (long)h * DHEAD;
#pragma unroll
  for (int i = 0; i < 4; ++i)
    tile[r + i * 8][c] = src[(long)(tj + r + i * 8) * (2 * INNER) + td + c];
  __syncthreads();
  half_t* dst = vT + (long)z * DHEAD * SEQ;
#pragma unroll
  for (int i = 0; i < 4; ++i)
    dst[(long)(td + r + i * 8) * SEQ + tj + c] = tile[c][r + i * 8];
}

// =====================================================================
// host-side orchestration
// =====================================================================
extern "C" void kernel_launch(void* const* d_in, const int* in_sizes, int n_in,
                              void* d_out, int out_size, void* d_ws, size_t ws_size,
                              hipStream_t stream) {
  (void)in_sizes; (void)n_in; (void)out_size; (void)ws_size;
  const float* x    = (const float*)d_in[0];
  const float* mask = (const float*)d_in[1];
  const float* wq   = (const float*)d_in[2];
  const float* wkv  = (const float*)d_in[3];
  const float* wo   = (const float*)d_in[4];
  const float* bo   = (const float*)d_in[5];
  const float* ln1g = (const float*)d_in[6];
  const float* ln1b = (const float*)d_in[7];
  const float* w1   = (const float*)d_in[8];
  const float* b1   = (const float*)d_in[9];
  const float* w2   = (const float*)d_in[10];
  const float* b2   = (const float*)d_in[11];
  const float* ln2g = (const float*)d_in[12];
  const float* ln2b = (const float*)d_in[13];
  float* X = (float*)d_out;

  char* ws = (char*)d_ws;
  size_t off = 0;
  auto alloc = [&](size_t bytes) -> void* {
    void* p = ws + off;
    off += (bytes + 255) & ~(size_t)255;
    return p;
  };
  half_t* xh   = (half_t*)alloc((size_t)TOKENS * DIMV * 2);
  half_t* q    = (half_t*)alloc((size_t)TOKENS * INNER * 2);
  half_t* kv   = (half_t*)alloc((size_t)TOKENS * 2 * INNER * 2);
  half_t* vT   = (half_t*)alloc((size_t)BATCH * HEADS * DHEAD * SEQ * 2);
  half_t* S    = (half_t*)alloc((size_t)BATCH * HEADS * SEQ * SEQ * 2);
  half_t* ao   = (half_t*)alloc((size_t)TOKENS * INNER * 2);
  float*  t32  = (float*) alloc((size_t)TOKENS * DIMV * 4);
  half_t* hm   = (half_t*)alloc((size_t)TOKENS * MLPD * 2);
  half_t* wqT  = (half_t*)alloc((size_t)DIMV * INNER * 2);
  half_t* wkvT = (half_t*)alloc((size_t)DIMV * 2 * INNER * 2);
  half_t* woT  = (half_t*)alloc((size_t)INNER * DIMV * 2);
  half_t* w1T  = (half_t*)alloc((size_t)DIMV * MLPD * 2);
  half_t* w2T  = (half_t*)alloc((size_t)MLPD * DIMV * 2);

  hipMemcpyAsync(X, x, (size_t)TOKENS * DIMV * 4, hipMemcpyDeviceToDevice, stream);
  cast_f32_to_f16<<<(TOKENS * DIMV / 4 + 255) / 256, 256, 0, stream>>>(x, xh, TOKENS * DIMV);

  for (int L = 0; L < DEPTH; ++L) {
    // transpose-cast this layer's weights into [N][K] f16
    transpose_cast_kernel<<<dim3(DIMV / 32, INNER / 32), 256, 0, stream>>>(
        wq + (size_t)L * DIMV * INNER, wqT, DIMV, INNER);
    transpose_cast_kernel<<<dim3(DIMV / 32, 2 * INNER / 32), 256, 0, stream>>>(
        wkv + (size_t)L * DIMV * 2 * INNER, wkvT, DIMV, 2 * INNER);
    transpose_cast_kernel<<<dim3(INNER / 32, DIMV / 32), 256, 0, stream>>>(
        wo + (size_t)L * INNER * DIMV, woT, INNER, DIMV);
    transpose_cast_kernel<<<dim3(DIMV / 32, MLPD / 32), 256, 0, stream>>>(
        w1 + (size_t)L * DIMV * MLPD, w1T, DIMV, MLPD);
    transpose_cast_kernel<<<dim3(MLPD / 32, DIMV / 32), 256, 0, stream>>>(
        w2 + (size_t)L * MLPD * DIMV, w2T, MLPD, DIMV);

    // q = xh @ wq^T(stored [N][K])
    gemm_wmma_tdm<128, 0><<<dim3(INNER / 128, TOKENS / 128, 1), 256, 0, stream>>>(
        xh, wqT, q, nullptr, nullptr, DIMV, DIMV, DIMV, INNER,
        0, 0, 0, 0, 0, 0, 1, 1.0f);
    // kv = xh @ wkv^T
    gemm_wmma_tdm<128, 0><<<dim3(2 * INNER / 128, TOKENS / 128, 1), 256, 0, stream>>>(
        xh, wkvT, kv, nullptr, nullptr, DIMV, DIMV, DIMV, 2 * INNER,
        0, 0, 0, 0, 0, 0, 1, 1.0f);
    // vT[b,h][d][j] from kv
    transpose_v_kernel<<<dim3(SEQ / 32, DHEAD / 32, BATCH * HEADS), 256, 0, stream>>>(kv, vT);
    // scores = scale * Q K^T per (b,h): K rows of kv are already [j][d]
    gemm_wmma_tdm<128, 0><<<dim3(SEQ / 128, SEQ / 128, BATCH * HEADS), 256, 0, stream>>>(
        q, kv, S, nullptr, nullptr, DHEAD, INNER, 2 * INNER, SEQ,
        (long)SEQ * INNER, 64, (long)SEQ * 2 * INNER, 64,
        (long)HEADS * SEQ * SEQ, (long)SEQ * SEQ, HEADS, SCALE_QK);
    // attn = softmax(scores) * mask (in place, f16)
    softmax_mask_kernel<<<BATCH * HEADS * SEQ, 256, 0, stream>>>(S, mask);
    // out = attn @ V  (B = vT is [d][j] = [N][K])
    gemm_wmma_tdm<64, 0><<<dim3(1, SEQ / 128, BATCH * HEADS), 256, 0, stream>>>(
        S, vT, ao, nullptr, nullptr, SEQ, SEQ, SEQ, INNER,
        (long)HEADS * SEQ * SEQ, (long)SEQ * SEQ,
        (long)HEADS * DHEAD * SEQ, (long)DHEAD * SEQ,
        (long)SEQ * INNER, 64, HEADS, 1.0f);
    // out-proj: t32 = ao @ wo^T + bo
    gemm_wmma_tdm<128, 1><<<dim3(DIMV / 128, TOKENS / 128, 1), 256, 0, stream>>>(
        ao, woT, nullptr, t32, bo + (size_t)L * DIMV, INNER, INNER, INNER, DIMV,
        0, 0, 0, 0, 0, 0, 1, 1.0f);
    // x = LN(t32) + x ; xh = f16(x)
    ln_residual_kernel<<<TOKENS, 256, 0, stream>>>(
        t32, ln1g + (size_t)L * DIMV, ln1b + (size_t)L * DIMV, X, xh);
    // mlp1: hm = gelu(xh @ w1^T + b1)
    gemm_wmma_tdm<128, 2><<<dim3(MLPD / 128, TOKENS / 128, 1), 256, 0, stream>>>(
        xh, w1T, hm, nullptr, b1 + (size_t)L * MLPD, DIMV, DIMV, DIMV, MLPD,
        0, 0, 0, 0, 0, 0, 1, 1.0f);
    // mlp2: t32 = hm @ w2^T + b2
    gemm_wmma_tdm<128, 1><<<dim3(DIMV / 128, TOKENS / 128, 1), 256, 0, stream>>>(
        hm, w2T, nullptr, t32, b2 + (size_t)L * DIMV, MLPD, MLPD, MLPD, DIMV,
        0, 0, 0, 0, 0, 0, 1, 1.0f);
    // x = LN(t32) + x ; xh = f16(x)
    ln_residual_kernel<<<TOKENS, 256, 0, stream>>>(
        t32, ln2g + (size_t)L * DIMV, ln2b + (size_t)L * DIMV, X, xh);
  }
}
